// NCNPredictor_16174846836921
// MI455X (gfx1250) — matrix operand order repaired and checked by reading, not compiled, not run on hardware
//
#include <hip/hip_runtime.h>
#include <hip/hip_bf16.h>
#include <stdint.h>

#define NNODES 10000
#define EEDGES 8192
#define INCH   128
#define HID    256
#define MAXCN  512
#define SF     260   // f32 LDS row stride (stride%banks -> conflict free)
#define SB     264   // bf16 LDS row stride

typedef __attribute__((ext_vector_type(16))) __bf16 v16bf;
typedef __attribute__((ext_vector_type(8)))  float  v8f;
typedef __attribute__((ext_vector_type(4)))  float  f4;

union Frag {
    uint4 q[2];
    v16bf v;
};

static __device__ inline unsigned short f32_to_bf16(float f) {
    unsigned int u = __float_as_uint(f);
    unsigned int lsb = (u >> 16) & 1u;
    u += 0x7fffu + lsb;               // round-to-nearest-even
    return (unsigned short)(u >> 16);
}

// -----------------------------------------------------------------------------
// Pack f32 weight [K,N] (row-major) into bf16 pre-swizzled B-fragment layout:
// dst[((nb*KB + kb)*32 + lane)*16 + s]  where lane = ((k%32)/8%2)*16 + n%16,
// s = (k%32)/16*8 + k%8  (matches V_WMMA 16-bit B 32x16 VGPR layout).
// -----------------------------------------------------------------------------
__global__ void pack_w_kernel(const float* __restrict__ W,
                              unsigned short* __restrict__ dst,
                              int K, int Nn) {
    int t = blockIdx.x * blockDim.x + threadIdx.x;
    int total = K * Nn;
    if (t >= total) return;
    int k = t / Nn, n = t % Nn;
    int kb = k >> 5, kin = k & 31;
    int half = (kin >> 3) & 1, koff = (kin >> 4) * 8 + (kin & 7);
    int nb = n >> 4, nloc = n & 15;
    int KB = K >> 5;
    int lane = half * 16 + nloc;
    size_t di = (((size_t)(nb * KB + kb)) * 32 + lane) * 16 + koff;
    dst[di] = f32_to_bf16(W[(size_t)k * Nn + n]);
}

// -----------------------------------------------------------------------------
// Per-edge: common-neighbor row intersection (bandwidth-bound adj streaming with
// non-temporal loads so the 400MB adj doesn't thrash L2), compact hit list in
// LDS, then aggregate x (L2-resident, default RT caching) over common
// neighbors. Also computes x[i]*x[j]. Outputs bf16 activations for WMMA GEMMs.
// -----------------------------------------------------------------------------
__global__ __launch_bounds__(256) void edge_feats_kernel(
    const float* __restrict__ x, const float* __restrict__ adj,
    const int* __restrict__ tar,
    unsigned short* __restrict__ xcn_bf, unsigned short* __restrict__ xij_bf) {
    __shared__ int   s_cnt;
    __shared__ int   s_idx[MAXCN];
    __shared__ float s_w[MAXCN];

    int e = blockIdx.x;
    int tid = threadIdx.x;
    int i = tar[e];
    int j = tar[EEDGES + e];
    if (tid == 0) s_cnt = 0;
    __syncthreads();

    const f4* ai4 = (const f4*)(adj + (size_t)i * NNODES);
    const f4* aj4 = (const f4*)(adj + (size_t)j * NNODES);
    for (int n4 = tid; n4 < NNODES / 4; n4 += 256) {
        f4 a = __builtin_nontemporal_load(&ai4[n4]);   // streaming: TH=NT
        f4 b = __builtin_nontemporal_load(&aj4[n4]);
#pragma unroll
        for (int q = 0; q < 4; ++q) {
            float pa = a[q], pb = b[q];
            if (pa != 0.f && pb != 0.f) {
                int k = atomicAdd(&s_cnt, 1);
                if (k < MAXCN) { s_idx[k] = n4 * 4 + q; s_w[k] = pa * pb; }
            }
        }
    }
    __syncthreads();

    int cnt = s_cnt;
    if (cnt > MAXCN) cnt = MAXCN;
    if (tid < INCH) {
        float acc = 0.f;
        for (int t = 0; t < cnt; ++t)
            acc += s_w[t] * x[(size_t)s_idx[t] * INCH + tid];
        xcn_bf[(size_t)e * INCH + tid] = f32_to_bf16(acc);
        float v = x[(size_t)i * INCH + tid] * x[(size_t)j * INCH + tid];
        xij_bf[(size_t)e * INCH + tid] = f32_to_bf16(v);
    }
}

// -----------------------------------------------------------------------------
// Fused MLP: Linear(K->256) -> LayerNorm -> ReLU -> (Linear(256->256) | dot 256->1)
// One 512-thread block (16 waves) per 16-row tile. Wave w owns column tile w.
// mode 0: cn branch, write f32
// mode 1: ij branch, out = hcn*BETA + result, write bf16 activation
// mode 2: final,     second linear is 256->1 dot, write d_out
// -----------------------------------------------------------------------------
__global__ __launch_bounds__(512) void mlp_kernel(
    const unsigned short* __restrict__ act, int K,
    const unsigned short* __restrict__ w1p, const float* __restrict__ b1,
    const float* __restrict__ g, const float* __restrict__ be,
    const unsigned short* __restrict__ w2p, const float* __restrict__ b2,
    const float* __restrict__ hcn_in, float* __restrict__ hcn_out,
    unsigned short* __restrict__ hbf_out,
    const float* __restrict__ w2l, const float* __restrict__ b2l,
    float* __restrict__ final_out, int mode) {
    __shared__ float          lds_f[16 * SF];
    __shared__ unsigned short lds_h[16 * SB];

    int tid  = threadIdx.x;
    int lane = tid & 31;
    int w    = tid >> 5;          // wave id 0..15 == output column block
    int eb   = blockIdx.x;        // 16-row block index
    int rowL = lane & 15;
    int half = lane >> 4;
    int KB   = K >> 5;

    // ---- GEMM1: h = act @ W1 ----
    v8f acc = {};
    const unsigned short* arow = act + (size_t)(eb * 16 + rowL) * K;
    for (int kb = 0; kb < KB; ++kb) {
        Frag a, b;
        a.q[0] = *(const uint4*)(arow + kb * 32 + half * 8);        // K = kb*32 + half*8 + [0..7]
        a.q[1] = *(const uint4*)(arow + kb * 32 + 16 + half * 8);   // K = kb*32 + 16 + half*8 + [0..7]
        const uint4* bp = (const uint4*)(w1p + (((size_t)(w * KB + kb)) * 32 + lane) * 16);
        b.q[0] = bp[0];
        b.q[1] = bp[1];
        acc = __builtin_amdgcn_wmma_f32_16x16x32_bf16(false, a.v, false, b.v,
                                                      (short)0, acc, false, false);
    }
    {   // bias + stage to LDS f32 (C layout: lane holds col, vgpr holds row)
        int col = w * 16 + rowL;
        float bias = b1[col];
#pragma unroll
        for (int vg = 0; vg < 8; ++vg) {
            int r = half * 8 + vg;
            lds_f[r * SF + col] = acc[vg] + bias;
        }
    }
    __syncthreads();

    // ---- LayerNorm + ReLU: wave w handles row w (256 cols, 8/lane) ----
    {
        float vals[8], s = 0.f, sq = 0.f;
#pragma unroll
        for (int k2 = 0; k2 < 8; ++k2) {
            float v = lds_f[w * SF + lane * 8 + k2];
            vals[k2] = v; s += v; sq += v * v;
        }
#pragma unroll
        for (int off = 16; off >= 1; off >>= 1) {
            s  += __shfl_xor(s,  off, 32);
            sq += __shfl_xor(sq, off, 32);
        }
        float mu  = s  * (1.0f / 256.0f);
        float var = sq * (1.0f / 256.0f) - mu * mu;
        float rs  = rsqrtf(var + 1e-5f);
#pragma unroll
        for (int k2 = 0; k2 < 8; ++k2) {
            int col = lane * 8 + k2;
            float v = (vals[k2] - mu) * rs * g[col] + be[col];
            v = fmaxf(v, 0.f);
            if (mode == 2) lds_f[w * SF + col] = v;
            else           lds_h[w * SB + col] = f32_to_bf16(v);
        }
    }
    __syncthreads();

    if (mode == 2) {
        // ---- final Linear(256 -> 1): per-row dot with w2l ----
        float s = 0.f;
#pragma unroll
        for (int k2 = 0; k2 < 8; ++k2) {
            int col = lane * 8 + k2;
            s += lds_f[w * SF + col] * w2l[col];
        }
#pragma unroll
        for (int off = 16; off >= 1; off >>= 1)
            s += __shfl_xor(s, off, 32);
        if (lane == 0) final_out[eb * 16 + w] = s + b2l[0];
        return;
    }

    // ---- GEMM2: out = relu(norm) @ W2  (K2 = 256) ----
    v8f acc2 = {};
    for (int kb = 0; kb < 8; ++kb) {
        Frag a, b;
        a.q[0] = *(const uint4*)&lds_h[rowL * SB + kb * 32 + half * 8];
        a.q[1] = *(const uint4*)&lds_h[rowL * SB + kb * 32 + 16 + half * 8];
        const uint4* bp = (const uint4*)(w2p + (((size_t)(w * 8 + kb)) * 32 + lane) * 16);
        b.q[0] = bp[0];
        b.q[1] = bp[1];
        acc2 = __builtin_amdgcn_wmma_f32_16x16x32_bf16(false, a.v, false, b.v,
                                                       (short)0, acc2, false, false);
    }
    {
        int col = w * 16 + rowL;
        float bias = b2[col];
#pragma unroll
        for (int vg = 0; vg < 8; ++vg) {
            int r = half * 8 + vg;
            int e = eb * 16 + r;
            float v = acc2[vg] + bias;
            if (mode == 0) {
                hcn_out[(size_t)e * HID + col] = v;
            } else {
                v = hcn_in[(size_t)e * HID + col] * 1.0f /*BETA*/ + v;
                hbf_out[(size_t)e * HID + col] = f32_to_bf16(v);
            }
        }
    }
}

// -----------------------------------------------------------------------------
extern "C" void kernel_launch(void* const* d_in, const int* in_sizes, int n_in,
                              void* d_out, int out_size, void* d_ws, size_t ws_size,
                              hipStream_t stream) {
    const float* x     = (const float*)d_in[0];
    const float* adj   = (const float*)d_in[1];
    const int*   tar   = (const int*)d_in[2];
    const float* w1_cn = (const float*)d_in[3];
    const float* b1_cn = (const float*)d_in[4];
    const float* g_cn  = (const float*)d_in[5];
    const float* be_cn = (const float*)d_in[6];
    const float* w2_cn = (const float*)d_in[7];
    const float* b2_cn = (const float*)d_in[8];
    const float* w1_ij = (const float*)d_in[9];
    const float* b1_ij = (const float*)d_in[10];
    const float* g_ij  = (const float*)d_in[11];
    const float* be_ij = (const float*)d_in[12];
    const float* w2_ij = (const float*)d_in[13];
    const float* b2_ij = (const float*)d_in[14];
    const float* w1_l  = (const float*)d_in[15];
    const float* b1_l  = (const float*)d_in[16];
    const float* g_l   = (const float*)d_in[17];
    const float* be_l  = (const float*)d_in[18];
    const float* w2_l  = (const float*)d_in[19];
    const float* b2_l  = (const float*)d_in[20];

    // workspace carve-out (256B aligned)
    char* ws = (char*)d_ws;
    size_t off = 0;
    auto nxt = [&](size_t bytes) -> void* {
        void* p = ws + off;
        off = (off + bytes + 255) & ~(size_t)255;
        return p;
    };
    unsigned short* w1cn_p = (unsigned short*)nxt((size_t)INCH * HID * 2);
    unsigned short* w2cn_p = (unsigned short*)nxt((size_t)HID * HID * 2);
    unsigned short* w1ij_p = (unsigned short*)nxt((size_t)INCH * HID * 2);
    unsigned short* w2ij_p = (unsigned short*)nxt((size_t)HID * HID * 2);
    unsigned short* w1l_p  = (unsigned short*)nxt((size_t)HID * HID * 2);
    unsigned short* xcn_bf = (unsigned short*)nxt((size_t)EEDGES * INCH * 2);
    unsigned short* xij_bf = (unsigned short*)nxt((size_t)EEDGES * INCH * 2);
    float*          hcn    = (float*)nxt((size_t)EEDGES * HID * 4);
    unsigned short* hbf    = (unsigned short*)nxt((size_t)EEDGES * HID * 2);

    // 1) prepack weights into WMMA B-fragment bf16 layout
    pack_w_kernel<<<(INCH * HID + 255) / 256, 256, 0, stream>>>(w1_cn, w1cn_p, INCH, HID);
    pack_w_kernel<<<(HID * HID + 255) / 256, 256, 0, stream>>>(w2_cn, w2cn_p, HID, HID);
    pack_w_kernel<<<(INCH * HID + 255) / 256, 256, 0, stream>>>(w1_ij, w1ij_p, INCH, HID);
    pack_w_kernel<<<(HID * HID + 255) / 256, 256, 0, stream>>>(w2_ij, w2ij_p, HID, HID);
    pack_w_kernel<<<(HID * HID + 255) / 256, 256, 0, stream>>>(w1_l, w1l_p, HID, HID);

    // 2) edge features: common-neighbor aggregation + x[i]*x[j]
    edge_feats_kernel<<<EEDGES, 256, 0, stream>>>(x, adj, tar, xcn_bf, xij_bf);

    // 3) fused MLPs (WMMA bf16 + in-LDS LayerNorm)
    mlp_kernel<<<EEDGES / 16, 512, 0, stream>>>(xcn_bf, INCH, w1cn_p, b1_cn, g_cn, be_cn,
                                                w2cn_p, b2_cn, nullptr, hcn, nullptr,
                                                nullptr, nullptr, nullptr, 0);
    mlp_kernel<<<EEDGES / 16, 512, 0, stream>>>(xij_bf, INCH, w1ij_p, b1_ij, g_ij, be_ij,
                                                w2ij_p, b2_ij, hcn, nullptr, hbf,
                                                nullptr, nullptr, nullptr, 1);
    mlp_kernel<<<EEDGES / 16, 512, 0, stream>>>(hbf, HID, w1l_p, b1_l, g_l, be_l,
                                                nullptr, nullptr, nullptr, nullptr, nullptr,
                                                w2_l, b2_l, (float*)d_out, 2);
}
